// GAT_36197984371175
// MI455X (gfx1250) — matrix operand order repaired and checked by reading, not compiled, not run on hardware
//
#include <hip/hip_runtime.h>
#include <cstdint>

// ---------------- problem constants (match reference) ----------------
#define NNODES      50000
#define NEDGES      800000
#define IN_C        128
#define HID         64
#define HEADS       4
#define NUM_GRAPHS  64
#define NUM_CLASSES 10
#define EPS_BN      1e-5f
#define NEG_SLOPE   0.2f

typedef __attribute__((ext_vector_type(16))) __bf16 v16bf;
typedef __attribute__((ext_vector_type(8)))  float  v8f;

// ---------------- WMMA GEMM: C[M,Ncols] = A[M,K] @ W[K,Ncols] ----------------
// fp32 in/out, bf16 matrix compute, f32 accumulate via v_wmma_f32_16x16x32_bf16.
// Double-buffered LDS K-loop: prefetch tile k+1 into registers while computing
// tile k, so the global-load wait lands after the WMMA quartet. A tile row-major
// in LDS, W tile transposed so both fragments are contiguous 2x ds_load_b128.
#define BM 128
#define BN 64
#define BK 32
#define APAD 8   // halves; 80B row stride keeps 16-lane column reads conflict-free

__launch_bounds__(256, 2)
__global__ void gemm_bf16_wmma(const float* __restrict__ A, const float* __restrict__ Wm,
                               float* __restrict__ Cm, int M, int K, int Ncols) {
  __shared__ __bf16 sA[2][BM][BK + APAD];    // sA[buf][m][k]
  __shared__ __bf16 sBt[2][BN][BK + APAD];   // sBt[buf][n][k] (transposed W tile)
  const int tid    = threadIdx.x;
  const int wave   = tid >> 5;       // 8 waves: wave = M-subtile (16 rows each)
  const int lane   = tid & 31;
  const int g      = lane >> 4;      // lane half (ISA 16-bit operand layout)
  const int lanelo = lane & 15;
  const int m0 = blockIdx.x * BM;
  const int n0 = blockIdx.y * BN;
  const int nkt = K / BK;

  v8f acc[4] = {};                   // four 16x16 f32 accumulators (n-subtiles)
  float4 aReg[4];                    // prefetch registers: A tile (4 quads/thread)
  float4 bReg[2];                    // prefetch registers: W tile (2 quads/thread)

  // global -> registers for K-tile kt
  auto gload = [&](int kt) {
    const int k0 = kt * BK;
    #pragma unroll
    for (int it = 0; it < 4; ++it) {
      int idx = tid + it * 256;
      int r = idx >> 3;              // BK/4 = 8 quads per row
      int q = idx & 7;
      int grow = m0 + r;
      aReg[it] = (float4){0.f, 0.f, 0.f, 0.f};
      if (grow < M) aReg[it] = *(const float4*)(A + (size_t)grow * K + k0 + q * 4);
    }
    #pragma unroll
    for (int it = 0; it < 2; ++it) {
      int idx = tid + it * 256;
      int r = idx >> 4;              // row k (BN/4 = 16 quads per row)
      int q = idx & 15;
      bReg[it] = *(const float4*)(Wm + (size_t)(k0 + r) * Ncols + n0 + q * 4);
    }
  };
  // registers -> LDS buffer buf (with f32 -> bf16 conversion)
  auto sstore = [&](int buf) {
    #pragma unroll
    for (int it = 0; it < 4; ++it) {
      int idx = tid + it * 256;
      int r = idx >> 3;
      int q = idx & 7;
      sA[buf][r][q * 4 + 0] = (__bf16)aReg[it].x;
      sA[buf][r][q * 4 + 1] = (__bf16)aReg[it].y;
      sA[buf][r][q * 4 + 2] = (__bf16)aReg[it].z;
      sA[buf][r][q * 4 + 3] = (__bf16)aReg[it].w;
    }
    #pragma unroll
    for (int it = 0; it < 2; ++it) {
      int idx = tid + it * 256;
      int r = idx >> 4;
      int q = idx & 15;
      sBt[buf][q * 4 + 0][r] = (__bf16)bReg[it].x;
      sBt[buf][q * 4 + 1][r] = (__bf16)bReg[it].y;
      sBt[buf][q * 4 + 2][r] = (__bf16)bReg[it].z;
      sBt[buf][q * 4 + 3][r] = (__bf16)bReg[it].w;
    }
  };

  gload(0);
  sstore(0);
  __syncthreads();

  const int mrow = wave * 16 + lanelo;
  for (int kt = 0; kt < nkt; ++kt) {
    const int cur = kt & 1;
    const int nxt = cur ^ 1;
    const bool more = (kt + 1) < nkt;
    if (more) gload(kt + 1);   // issue global loads; consumed after compute

    // A fragment, 16-bit A 16x32 layout: lanes 0-15 -> K {0..7,16..23},
    // lanes 16-31 -> K {8..15,24..31}; two contiguous 16B chunks per lane.
    v16bf afrag;
    #pragma unroll
    for (int i = 0; i < 16; ++i) {
      int kk = (i & 7) + ((i >> 3) << 4) + (g << 3);
      afrag[i] = sA[cur][mrow][kk];
    }
    // Preload all four B fragments (16-bit B 32x16 layout: lanes 0-15 -> K {0..7,16..23},
    // lanes 16-31 -> K {8..15,24..31}), then run the WMMA quartet back-to-back.
    v16bf bfrag[4];
    #pragma unroll
    for (int t = 0; t < 4; ++t) {
      const int ncol = t * 16 + lanelo;
      #pragma unroll
      for (int i = 0; i < 16; ++i) bfrag[t][i] = sBt[cur][ncol][(i & 7) + ((i >> 3) << 4) + (g << 3)];
    }
    #pragma unroll
    for (int t = 0; t < 4; ++t) {
      acc[t] = __builtin_amdgcn_wmma_f32_16x16x32_bf16(
          false, afrag, false, bfrag[t], (short)0, acc[t], false, false);
    }

    if (more) sstore(nxt);     // waits on the prefetch loads here, not earlier
    __syncthreads();
  }

  // C/D layout: VGPR v holds row m = v + 8*(lane/16), col n = lane%16
  #pragma unroll
  for (int t = 0; t < 4; ++t) {
    const int gc = n0 + t * 16 + lanelo;
    #pragma unroll
    for (int v = 0; v < 8; ++v) {
      const int gr = m0 + wave * 16 + v + 8 * g;
      if (gr < M) Cm[(size_t)gr * Ncols + gc] = acc[t][v];
    }
  }
}

// ---------------- helpers ----------------
__global__ void fill_f32(float* __restrict__ p, float v, size_t n) {
  size_t i = (size_t)blockIdx.x * blockDim.x + threadIdx.x;
  if (i < n) p[i] = v;
}

// per-node attention logits: as_[n,h] = <h[n,h,:], a_src[h,:]>, same for a_dst
__global__ void node_logits(const float* __restrict__ h, const float* __restrict__ asr,
                            const float* __restrict__ adr, float* __restrict__ as_,
                            float* __restrict__ ad_, int H) {
  int idx = blockIdx.x * blockDim.x + threadIdx.x;
  if (idx >= NNODES * H) return;
  int node = idx / H, head = idx % H;
  const float* hp = h + (size_t)node * H * HID + (size_t)head * HID;
  const float* ap = asr + head * HID;
  const float* bp = adr + head * HID;
  float s1 = 0.f, s2 = 0.f;
  #pragma unroll 4
  for (int c = 0; c < HID; ++c) {
    float v = hp[c];
    s1 += v * ap[c];
    s2 += v * bp[c];
  }
  as_[idx] = s1;
  ad_[idx] = s2;
}

// e = leaky_relu(as_[src] + ad_[dst]); segment max over dst (float atomic max)
__global__ void edge_logits_max(const int* __restrict__ src, const int* __restrict__ dst,
                                const float* __restrict__ as_, const float* __restrict__ ad_,
                                float* __restrict__ e, float* __restrict__ emax, int H) {
  int i = blockIdx.x * blockDim.x + threadIdx.x;
  if (i >= NEDGES * H) return;
  int eid = i / H, head = i % H;
  int s = src[eid], d = dst[eid];
  float v = as_[(size_t)s * H + head] + ad_[(size_t)d * H + head];
  v = v > 0.f ? v : NEG_SLOPE * v;
  e[i] = v;
  __hip_atomic_fetch_max(&emax[(size_t)d * H + head], v, __ATOMIC_RELAXED,
                         __HIP_MEMORY_SCOPE_AGENT);
}

// ex = exp(e - emax[dst]) (in place); denom[dst] += ex
__global__ void edge_exp_sum(const int* __restrict__ dst, float* __restrict__ e,
                             const float* __restrict__ emax, float* __restrict__ denom, int H) {
  int i = blockIdx.x * blockDim.x + threadIdx.x;
  if (i >= NEDGES * H) return;
  int eid = i / H, head = i % H;
  int d = dst[eid];
  float v = __expf(e[i] - emax[(size_t)d * H + head]);
  e[i] = v;
  atomicAdd(&denom[(size_t)d * H + head], v);
}

// out[dst] += h[src] * alpha  (one wave per edge; lane strides channels)
__global__ void edge_aggregate(const int* __restrict__ src, const int* __restrict__ dst,
                               const float* __restrict__ h, const float* __restrict__ ex,
                               const float* __restrict__ denom, float* __restrict__ out, int H) {
  int gid = blockIdx.x * blockDim.x + threadIdx.x;
  int eid = gid >> 5;
  int lane = gid & 31;
  if (eid >= NEDGES) return;
  int s = src[eid], d = dst[eid];
  const int HC = H * HID;
  for (int c = lane; c < HC; c += 32) {
    int head = c >> 6;  // HID == 64 for every layer
    float alpha = ex[(size_t)eid * H + head] / denom[(size_t)d * H + head];
    atomicAdd(&out[(size_t)d * HC + c], h[(size_t)s * HC + c] * alpha);
  }
}

// x = relu(x + bias) in place; accumulate per-channel sum / sum-of-squares
__global__ void bias_relu_bnsum(float* __restrict__ x, const float* __restrict__ b,
                                float* __restrict__ bsum, float* __restrict__ bsq, int C) {
  int c = threadIdx.x;  // blockDim.x == C
  float bias = b[c], s = 0.f, q = 0.f;
  for (int r = blockIdx.x; r < NNODES; r += gridDim.x) {
    size_t idx = (size_t)r * C + c;
    float v = x[idx] + bias;
    v = fmaxf(v, 0.f);
    x[idx] = v;
    s += v;
    q += v * v;
  }
  atomicAdd(&bsum[c], s);
  atomicAdd(&bsq[c], q);
}

// fold BN into per-channel scale/shift (biased variance, like jnp.var)
__global__ void bn_finalize(float* __restrict__ bsum, float* __restrict__ bsq,
                            const float* __restrict__ gamma, const float* __restrict__ beta,
                            int C) {
  int c = threadIdx.x;
  if (c >= C) return;
  float mean  = bsum[c] / (float)NNODES;
  float var   = bsq[c] / (float)NNODES - mean * mean;
  float scale = gamma[c] * rsqrtf(var + EPS_BN);
  bsum[c] = scale;
  bsq[c]  = beta[c] - mean * scale;
}

__global__ void bn_apply(float* __restrict__ x, const float* __restrict__ scale,
                         const float* __restrict__ shift, size_t n, int C) {
  size_t i = (size_t)blockIdx.x * blockDim.x + threadIdx.x;
  if (i >= n) return;
  int c = (int)(i % (size_t)C);
  x[i] = x[i] * scale[c] + shift[c];
}

// global mean pool (sum + counts)
__global__ void pool_sum(const float* __restrict__ h, const int* __restrict__ batch,
                         float* __restrict__ pool, float* __restrict__ cnt) {
  int i = blockIdx.x * blockDim.x + threadIdx.x;
  if (i >= NNODES * HID) return;
  int node = i / HID, c = i % HID;
  int gph = batch[node];
  atomicAdd(&pool[gph * HID + c], h[i]);
  if (c == 0) atomicAdd(&cnt[gph], 1.f);
}

__global__ void final_fc(const float* __restrict__ pool, const float* __restrict__ cnt,
                         const float* __restrict__ fcW, const float* __restrict__ fcb,
                         float* __restrict__ out) {
  int i = blockIdx.x * blockDim.x + threadIdx.x;
  if (i >= NUM_GRAPHS * NUM_CLASSES) return;
  int gph = i / NUM_CLASSES, k = i % NUM_CLASSES;
  float inv = 1.f / fmaxf(cnt[gph], 1.f);
  float s = fcb[k];
  for (int c = 0; c < HID; ++c) s += pool[gph * HID + c] * inv * fcW[c * NUM_CLASSES + k];
  out[i] = s;
}

// ---------------- host orchestration ----------------
extern "C" void kernel_launch(void* const* d_in, const int* in_sizes, int n_in,
                              void* d_out, int out_size, void* d_ws, size_t ws_size,
                              hipStream_t stream) {
  (void)in_sizes; (void)n_in; (void)out_size; (void)ws_size;
  const float* x    = (const float*)d_in[0];
  const int*   eidx = (const int*)d_in[1];
  const int*   bat  = (const int*)d_in[2];
  const float* W1   = (const float*)d_in[3];
  const float* as1  = (const float*)d_in[4];
  const float* ad1  = (const float*)d_in[5];
  const float* b1   = (const float*)d_in[6];
  const float* g1   = (const float*)d_in[7];
  const float* be1  = (const float*)d_in[8];
  const float* W2   = (const float*)d_in[9];
  const float* as2  = (const float*)d_in[10];
  const float* ad2  = (const float*)d_in[11];
  const float* b2   = (const float*)d_in[12];
  const float* g2   = (const float*)d_in[13];
  const float* be2  = (const float*)d_in[14];
  const float* W3   = (const float*)d_in[15];
  const float* as3  = (const float*)d_in[16];
  const float* ad3  = (const float*)d_in[17];
  const float* b3   = (const float*)d_in[18];
  const float* g3   = (const float*)d_in[19];
  const float* be3  = (const float*)d_in[20];
  const float* fcW  = (const float*)d_in[21];
  const float* fcb  = (const float*)d_in[22];
  const int* src = eidx;
  const int* dst = eidx + NEDGES;

  // workspace carve-up (floats)
  float* ws = (float*)d_ws;
  size_t o = 0;
  float* bufH  = ws + o; o += (size_t)NNODES * 256;       // GEMM output h (pre-attention)
  float* bufA  = ws + o; o += (size_t)NNODES * 256;       // layer outputs (ping)
  float* bufB  = ws + o; o += (size_t)NNODES * 256;       // layer outputs (pong)
  float* asb   = ws + o; o += (size_t)NNODES * HEADS;
  float* adb   = ws + o; o += (size_t)NNODES * HEADS;
  float* eEx   = ws + o; o += (size_t)NEDGES * HEADS;     // e, then ex (in place)
  float* emax  = ws + o; o += (size_t)NNODES * HEADS;
  float* denom = ws + o; o += (size_t)NNODES * HEADS;
  float* bsum  = ws + o; o += 256;
  float* bsq   = ws + o; o += 256;
  float* pool  = ws + o; o += (size_t)NUM_GRAPHS * HID;
  float* cnt   = ws + o; o += NUM_GRAPHS;

  auto layer = [&](const float* in, int Kin, const float* Wm, const float* asr,
                   const float* adr, const float* bias, const float* gamma,
                   const float* beta, int H, float* out) {
    const int HC = H * HID;
    dim3 gg((NNODES + BM - 1) / BM, HC / BN);
    gemm_bf16_wmma<<<gg, 256, 0, stream>>>(in, Wm, bufH, NNODES, Kin, HC);
    node_logits<<<(NNODES * H + 255) / 256, 256, 0, stream>>>(bufH, asr, adr, asb, adb, H);
    fill_f32<<<(NNODES * H + 255) / 256, 256, 0, stream>>>(emax, -1e30f, (size_t)NNODES * H);
    fill_f32<<<(NNODES * H + 255) / 256, 256, 0, stream>>>(denom, 0.f, (size_t)NNODES * H);
    fill_f32<<<(unsigned)(((size_t)NNODES * HC + 255) / 256), 256, 0, stream>>>(
        out, 0.f, (size_t)NNODES * HC);
    edge_logits_max<<<(NEDGES * H + 255) / 256, 256, 0, stream>>>(src, dst, asb, adb, eEx, emax, H);
    edge_exp_sum<<<(NEDGES * H + 255) / 256, 256, 0, stream>>>(dst, eEx, emax, denom, H);
    edge_aggregate<<<(unsigned)(((size_t)NEDGES * 32 + 255) / 256), 256, 0, stream>>>(
        src, dst, bufH, eEx, denom, out, H);
    fill_f32<<<1, 256, 0, stream>>>(bsum, 0.f, (size_t)HC);
    fill_f32<<<1, 256, 0, stream>>>(bsq, 0.f, (size_t)HC);
    bias_relu_bnsum<<<512, HC, 0, stream>>>(out, bias, bsum, bsq, HC);
    bn_finalize<<<1, HC, 0, stream>>>(bsum, bsq, gamma, beta, HC);
    bn_apply<<<(unsigned)(((size_t)NNODES * HC + 255) / 256), 256, 0, stream>>>(
        out, bsum, bsq, (size_t)NNODES * HC, HC);
  };

  layer(x,    IN_C,        W1, as1, ad1, b1, g1, be1, HEADS, bufA);
  layer(bufA, HEADS * HID, W2, as2, ad2, b2, g2, be2, HEADS, bufB);
  layer(bufB, HEADS * HID, W3, as3, ad3, b3, g3, be3, 1,     bufA);  // h3: N x 64

  fill_f32<<<(NUM_GRAPHS * HID + 255) / 256, 256, 0, stream>>>(pool, 0.f,
                                                               (size_t)NUM_GRAPHS * HID);
  fill_f32<<<1, 64, 0, stream>>>(cnt, 0.f, (size_t)NUM_GRAPHS);
  pool_sum<<<(NNODES * HID + 255) / 256, 256, 0, stream>>>(bufA, bat, pool, cnt);
  final_fc<<<(NUM_GRAPHS * NUM_CLASSES + 255) / 256, 256, 0, stream>>>(pool, cnt, fcW, fcb,
                                                                       (float*)d_out);
}
